// SwinAttention_66391604462522
// MI455X (gfx1250) — matrix operand order repaired and checked by reading, not compiled, not run on hardware
//
#include <hip/hip_runtime.h>

// ---------------------------------------------------------------------------
// SwinV2 window attention, fused per-window kernel for gfx1250 (MI455X).
// Precision: bf16 inputs to v_wmma_f32_16x16x32_bf16, fp32 accumulation.
// Roofline: op is HBM-bound (~270MB @ 23.3TB/s ~= 12us); bf16 WMMA puts the
// ~4.7M wmma issues below the memory floor (fp32 16x16x4 WMMA would be ~5x over).
// Data path: GLOBAL_LOAD_ASYNC_TO_LDS_B128 (ASYNCcnt) streams the rolled x
// tile into LDS; one conversion pass makes a bf16 tile so GEMM A-frags are
// two ds_load_b128 each. GEMMs are ni-outer with the 8 B-frags hoisted into
// VGPRs across the mi loop -> each weight byte is read from L2 exactly once
// per workgroup (qkv 384KB + proj 128KB), ~1GB total L2 traffic.
// Register budget: outer loops serialized (#pragma unroll 1) -> no spills.
// ---------------------------------------------------------------------------

typedef __attribute__((ext_vector_type(16))) __bf16 bf16x16;
typedef __attribute__((ext_vector_type(8)))  __bf16 bf16x8;
typedef __attribute__((ext_vector_type(8)))  float  f32x8;
typedef __attribute__((ext_vector_type(4)))  float  f32x4;
typedef __attribute__((ext_vector_type(4)))  int    i32x4;

#define WSZ     8
#define HEADS   8
#define CH      256
#define UNITS   32
#define SHIFT   4
#define TOK     64
#define EPS     1.55e-5f
#define LOG100  4.605170185988091f

#define WMMA_BF16(A, B, Cacc) \
  __builtin_amdgcn_wmma_f32_16x16x32_bf16(false, (A), false, (B), (short)0, (Cacc), false, false)

#if defined(__gfx1250__) && __has_builtin(__builtin_amdgcn_global_load_async_to_lds_b128)
#define HAVE_ASYNC_LDS 1
#endif

typedef __attribute__((address_space(1))) i32x4 g_i32x4;   // global <4 x i32>
typedef __attribute__((address_space(3))) i32x4 l_i32x4;   // LDS    <4 x i32>

__device__ __forceinline__ int seg_lab(int n) {       // Swin shift-mask segment label
  return n < 248 ? 0 : (n < 252 ? 1 : 2);
}

// ---------------------------------------------------------------------------
// Prep 1: transpose + bf16-convert weights so WMMA B-frags read contiguous K.
// ---------------------------------------------------------------------------
__global__ void prep_transpose(const float* __restrict__ qkv_w,
                               const float* __restrict__ proj_w,
                               __bf16* __restrict__ qkv_wT,
                               __bf16* __restrict__ proj_wT) {
  int i = blockIdx.x * blockDim.x + threadIdx.x;
  if (i < 768 * 256) {
    int n = i >> 8, k = i & 255;
    qkv_wT[i] = (__bf16)qkv_w[k * 768 + n];
  } else {
    int j = i - 768 * 256;
    if (j < 256 * 256) {
      int n = j >> 8, k = j & 255;
      proj_wT[j] = (__bf16)proj_w[k * 256 + n];
    }
  }
}

// ---------------------------------------------------------------------------
// Prep 2: CPB MLP over the 225 log-spaced relative coords -> cpb[225][8]
// ---------------------------------------------------------------------------
__global__ void prep_cpb(const float* __restrict__ w1, const float* __restrict__ b1,
                         const float* __restrict__ w2, float* __restrict__ cpb) {
  int i = threadIdx.x;
  if (i >= 225) return;
  int dh = i / 15 - 7, dw = i % 15 - 7;
  auto logc = [](int d) {
    float t = (float)d * (8.0f / 7.0f);
    float s = (t > 0.f) ? 1.f : (t < 0.f ? -1.f : 0.f);
    return s * log2f(1.f + fabsf(t)) * (1.0f / 3.0f);
  };
  float t0 = logc(dh), t1 = logc(dw);
  float acc[8];
#pragma unroll
  for (int h = 0; h < 8; ++h) acc[h] = 0.f;
  for (int j = 0; j < 512; ++j) {
    float hid = fmaxf(t0 * w1[j] + t1 * w1[512 + j] + b1[j], 0.f);
#pragma unroll
    for (int h = 0; h < 8; ++h) acc[h] += hid * w2[j * 8 + h];
  }
#pragma unroll
  for (int h = 0; h < 8; ++h) cpb[i * 8 + h] = acc[h];
}

// ---------------------------------------------------------------------------
// Prep 3: expand bias_g[h][64][64] = 16*sigmoid(cpb[REL_INDEX])
// ---------------------------------------------------------------------------
__global__ void prep_bias(const float* __restrict__ cpb, float* __restrict__ bias_g) {
  int e = blockIdx.x * 256 + threadIdx.x;
  if (e >= 8 * 64 * 64) return;
  int h = e >> 12, q = (e >> 6) & 63, k = e & 63;
  int rel = ((q >> 3) - (k >> 3) + 7) * 15 + ((q & 7) - (k & 7) + 7);
  float c = cpb[rel * 8 + h];
  bias_g[e] = 16.f / (1.f + __expf(-c));
}

// ---------------------------------------------------------------------------
// Main fused kernel: one workgroup (8 wave32) per 8x8 window.
// ---------------------------------------------------------------------------
__global__ __launch_bounds__(256)
void swin_window_kernel(const float* __restrict__ x,
                        const __bf16* __restrict__ qkv_wT,
                        const __bf16* __restrict__ proj_wT,
                        const float* __restrict__ q_bias,
                        const float* __restrict__ v_bias,
                        const float* __restrict__ scale,
                        const float* __restrict__ bias_g,
                        const float* __restrict__ proj_b,
                        float* __restrict__ out) {
  __shared__ __bf16 s_q[TOK * CH];        // q (bf16); reused for attn-out in stage 3/4
  __shared__ __bf16 s_k[TOK * CH];        // k (bf16)
  __shared__ __bf16 s_vT[CH * TOK];       // v transposed: [unit 0..255][token 0..63]
  __shared__ float  s_prf[TOK * CH];      // 64KB: fp32 x landing + bf16 x; later probs

  __bf16* s_pr = (__bf16*)s_prf;          // probs alias [8 heads][64][64] bf16
  __bf16* s_x  = s_pr + TOK * CH / 2;     // bf16 x tile at bytes 16K..48K of s_prf

  const int tid  = threadIdx.x;
  const int lane = tid & 31;
  const int wv   = __builtin_amdgcn_readfirstlane(tid >> 5);  // SGPR: scalar branches
  const int blk  = blockIdx.x;            // b*1024 + wh*32 + ww
  const int b    = blk >> 10;
  const int wh   = (blk >> 5) & 31;
  const int ww   = blk & 31;
  const int cl   = lane & 15;             // tile row/column lane index
  const int kh   = lane >> 4;             // K-half select (contiguous k = 16*kh + s)

  // Warm L2 for shared tables (global_prefetch_b8; probe-confirmed lowering).
  __builtin_prefetch(qkv_wT + (size_t)(6 * wv) * 16 * CH, 0, 1);
  __builtin_prefetch(proj_wT + (size_t)wv * 32 * CH, 0, 1);
  __builtin_prefetch(bias_g + wv * 4096, 0, 1);

  const int r  = tid >> 2;                // window token 0..63 owned by this thread
  const int qd = tid & 3;                 // channel quarter (64 values)

  // ---- Stage 0: rolled x tile -> LDS, convert once to bf16 -------------------
  {
    const int gh = (wh * WSZ + (r >> 3) + SHIFT) & 255;
    const int gw = (ww * WSZ + (r & 7) + SHIFT) & 255;
    const float* src = x + ((size_t)((b * 256 + gh) * 256 + gw)) * CH + qd * 64;
#ifdef HAVE_ASYNC_LDS
    // fp32 landing via async copy (no VGPR round-trip), then one convert pass.
    float* dstf = s_prf + r * CH + qd * 64;
    g_i32x4* g = (g_i32x4*)(unsigned long long)(__UINTPTR_TYPE__)src;
    l_i32x4* l = (l_i32x4*)(unsigned)(__UINTPTR_TYPE__)dstf;
#pragma unroll
    for (int i = 0; i < 16; ++i)
      __builtin_amdgcn_global_load_async_to_lds_b128(g + i, l + i, 0, 0);
#if __has_builtin(__builtin_amdgcn_s_wait_asynccnt)
    __builtin_amdgcn_s_wait_asynccnt(0);
#else
    asm volatile("s_wait_asynccnt 0x0" ::: "memory");
#endif
    __syncthreads();
    f32x4 regs[16];
#pragma unroll
    for (int i = 0; i < 16; ++i) regs[i] = *(const f32x4*)(dstf - qd * 0 + i * 4);
    __syncthreads();                       // all fp32 reads done before overwrite
    __bf16* dstb = s_x + r * CH + qd * 64;
#pragma unroll
    for (int i = 0; i < 8; ++i) {
      bf16x8 p;
#pragma unroll
      for (int j = 0; j < 4; ++j) {
        p[j]     = (__bf16)regs[2 * i][j];
        p[4 + j] = (__bf16)regs[2 * i + 1][j];
      }
      *(bf16x8*)(dstb + i * 8) = p;
    }
#else
    __bf16* dstb = s_x + r * CH + qd * 64;
#pragma unroll
    for (int i = 0; i < 8; ++i) {
      f32x4 u0 = *(const f32x4*)(src + i * 8 + 0);
      f32x4 u1 = *(const f32x4*)(src + i * 8 + 4);
      bf16x8 p;
#pragma unroll
      for (int j = 0; j < 4; ++j) { p[j] = (__bf16)u0[j]; p[4 + j] = (__bf16)u1[j]; }
      *(bf16x8*)(dstb + i * 8) = p;
    }
#endif
  }
  __syncthreads();

  // ---- Stage 1: qkv[64,768] = x_tile @ qkv_w + bias (WMMA bf16) --------------
  // ni outer: 8 B-frags loaded once from L2 and held in 64 VGPRs across the
  // mi loop; A-frags streamed from bf16 LDS (2 ds_load_b128 each).
  {
#pragma unroll 1
    for (int ni6 = 0; ni6 < 6; ++ni6) {
      const int ni = 6 * wv + ni6;                       // SGPR-uniform
      const int gn = ni * 16 + cl;                       // output column 0..767
      const float bval = gn < 256 ? q_bias[gn] : (gn < 512 ? 0.f : v_bias[gn - 512]);
      bf16x16 bfr[8];
#pragma unroll
      for (int kk = 0; kk < 8; ++kk)
        bfr[kk] = *(const bf16x16*)(qkv_wT + (size_t)gn * CH + kk * 32 + kh * 16);
#pragma unroll 1
      for (int mi = 0; mi < 4; ++mi) {
        f32x8 acc;
#pragma unroll
        for (int i = 0; i < 8; ++i) acc[i] = bval;
#pragma unroll
        for (int kk = 0; kk < 8; ++kk) {
          const __bf16* ap = s_x + (mi * 16 + cl) * CH + kk * 32 + kh * 16;
          bf16x8 lo = *(const bf16x8*)(ap);
          bf16x8 hi = *(const bf16x8*)(ap + 8);
          bf16x16 a;
#pragma unroll
          for (int j = 0; j < 8; ++j) { a[j] = lo[j]; a[8 + j] = hi[j]; }
          acc = WMMA_BF16(a, bfr[kk], acc);
        }
        // Epilogue: q/k/v region is uniform in ni -> scalar branch (SGPR wv).
        if (ni < 16) {
#pragma unroll
          for (int g = 0; g < 8; ++g)
            s_q[(mi * 16 + kh * 8 + g) * CH + gn] = (__bf16)acc[g];
        } else if (ni < 32) {
#pragma unroll
          for (int g = 0; g < 8; ++g)
            s_k[(mi * 16 + kh * 8 + g) * CH + (gn - 256)] = (__bf16)acc[g];
        } else {
#pragma unroll
          for (int g = 0; g < 8; ++g)
            s_vT[(gn - 512) * TOK + (mi * 16 + kh * 8 + g)] = (__bf16)acc[g];
        }
      }
    }
  }
  __syncthreads();

  // ---- Stage 2: per-(token,head) L2 norm of q,k; fold logit scale into q -----
  {
#pragma unroll
    for (int it = 0; it < 4; ++it) {
      const int task = tid + it * 256;                   // 0..1023
      const int h = task & 7;
      const int row = (task >> 3) & 63;
      const bool isq = task < 512;                       // constant per iteration
      __bf16* p = (isq ? s_q : s_k) + row * CH + h * UNITS;
      float vals[32];
      float ss = 0.f;
#pragma unroll
      for (int c4 = 0; c4 < 4; ++c4) {
        bf16x8 vv = *(const bf16x8*)(p + c4 * 8);
#pragma unroll
        for (int j = 0; j < 8; ++j) { float f = (float)vv[j]; vals[c4 * 8 + j] = f; ss += f * f; }
      }
      float rn = rsqrtf(fmaxf(ss, EPS));
      if (isq) rn *= __expf(fminf(scale[h], LOG100));
#pragma unroll
      for (int c4 = 0; c4 < 4; ++c4) {
        bf16x8 vv;
#pragma unroll
        for (int j = 0; j < 8; ++j) vv[j] = (__bf16)(vals[c4 * 8 + j] * rn);
        *(bf16x8*)(p + c4 * 8) = vv;
      }
    }
  }
  __syncthreads();

  // ---- Stage 3: attention, one head per wave ---------------------------------
  {
    const int h = wv;
    // k B-frags resident across both halves
    bf16x16 bk[4];
#pragma unroll
    for (int ni = 0; ni < 4; ++ni)
      bk[ni] = *(const bf16x16*)(s_k + (ni * 16 + cl) * CH + h * UNITS + kh * 16);

    const float* bh = bias_g + h * 4096;

    // QK^T + bias + mask + softmax, two M-tiles at a time (halves register peak)
#pragma unroll 1
    for (int half = 0; half < 2; ++half) {
      bf16x16 aq[2];
#pragma unroll
      for (int m2 = 0; m2 < 2; ++m2) {
        const int mi = half * 2 + m2;
        const __bf16* ap = s_q + (mi * 16 + cl) * CH + h * UNITS + kh * 16;
        bf16x8 lo = *(const bf16x8*)(ap);
        bf16x8 hi = *(const bf16x8*)(ap + 8);
#pragma unroll
        for (int j = 0; j < 8; ++j) { aq[m2][j] = lo[j]; aq[m2][8 + j] = hi[j]; }
      }
      f32x8 at[2][4];
#pragma unroll
      for (int m2 = 0; m2 < 2; ++m2)
#pragma unroll
        for (int ni = 0; ni < 4; ++ni) {
          f32x8 z;
#pragma unroll
          for (int i = 0; i < 8; ++i) z[i] = 0.f;
          at[m2][ni] = WMMA_BF16(aq[m2], bk[ni], z);
        }
#pragma unroll
      for (int m2 = 0; m2 < 2; ++m2)
#pragma unroll
        for (int ni = 0; ni < 4; ++ni)
#pragma unroll
          for (int g = 0; g < 8; ++g) {
            const int qrow = (half * 2 + m2) * 16 + kh * 8 + g;
            const int kcol = ni * 16 + cl;
            const int lq = seg_lab(wh * 8 + (qrow >> 3)) * 3 + seg_lab(ww * 8 + (qrow & 7));
            const int lk = seg_lab(wh * 8 + (kcol >> 3)) * 3 + seg_lab(ww * 8 + (kcol & 7));
            at[m2][ni][g] += bh[qrow * 64 + kcol] + ((lq == lk) ? 0.f : -100.f);
          }
#pragma unroll
      for (int m2 = 0; m2 < 2; ++m2)
#pragma unroll
        for (int g = 0; g < 8; ++g) {
          float m = fmaxf(fmaxf(at[m2][0][g], at[m2][1][g]), fmaxf(at[m2][2][g], at[m2][3][g]));
#pragma unroll
          for (int off = 1; off < 16; off <<= 1) m = fmaxf(m, __shfl_xor(m, off, 16));
          float p[4], s = 0.f;
#pragma unroll
          for (int ni = 0; ni < 4; ++ni) { p[ni] = __expf(at[m2][ni][g] - m); s += p[ni]; }
#pragma unroll
          for (int off = 1; off < 16; off <<= 1) s += __shfl_xor(s, off, 16);
          const float inv = 1.f / s;
          const int qrow = (half * 2 + m2) * 16 + kh * 8 + g;
#pragma unroll
          for (int ni = 0; ni < 4; ++ni)
            s_pr[h * 4096 + qrow * 64 + (ni * 16 + cl)] = (__bf16)(p[ni] * inv);
        }
    }

    // out_h[64,32] = probs @ v_h  (K=64 -> 2 bf16 WMMA steps), per-wave LDS slice
    f32x8 ov[4][2];
#pragma unroll
    for (int mi = 0; mi < 4; ++mi)
#pragma unroll
      for (int nj = 0; nj < 2; ++nj)
#pragma unroll
        for (int i = 0; i < 8; ++i) ov[mi][nj][i] = 0.f;

#pragma unroll 1
    for (int kt = 0; kt < 2; ++kt) {
      bf16x16 ap[4];
#pragma unroll
      for (int mi = 0; mi < 4; ++mi) {
        const __bf16* pp = s_pr + h * 4096 + (mi * 16 + cl) * 64 + kt * 32 + kh * 16;
        bf16x8 lo = *(const bf16x8*)(pp);
        bf16x8 hi = *(const bf16x8*)(pp + 8);
#pragma unroll
        for (int j = 0; j < 8; ++j) { ap[mi][j] = lo[j]; ap[mi][8 + j] = hi[j]; }
      }
      bf16x16 bv[2];
#pragma unroll
      for (int nj = 0; nj < 2; ++nj)
        bv[nj] = *(const bf16x16*)(s_vT + (h * 32 + nj * 16 + cl) * TOK + kt * 32 + kh * 16);
#pragma unroll
      for (int mi = 0; mi < 4; ++mi)
#pragma unroll
        for (int nj = 0; nj < 2; ++nj)
          ov[mi][nj] = WMMA_BF16(ap[mi], bv[nj], ov[mi][nj]);
    }

#pragma unroll
    for (int mi = 0; mi < 4; ++mi)
#pragma unroll
      for (int nj = 0; nj < 2; ++nj)
#pragma unroll
        for (int g = 0; g < 8; ++g)
          s_q[(mi * 16 + kh * 8 + g) * CH + (h * 32 + nj * 16 + cl)] = (__bf16)ov[mi][nj][g];
  }
  __syncthreads();

  // ---- Stage 4: proj [64,256] = attn_out @ proj_w + proj_b; reverse-roll store
  // nj outer: B-frags hoisted across the mi loop (proj weights read once/WG).
  {
    const __bf16* s_out = s_q;
#pragma unroll 1
    for (int nj2 = 0; nj2 < 2; ++nj2) {
      const int nj = 2 * wv + nj2;                       // SGPR-uniform, 0..15
      const int gn = nj * 16 + cl;
      const float pb = proj_b[gn];
      bf16x16 bfr[8];
#pragma unroll
      for (int kk = 0; kk < 8; ++kk)
        bfr[kk] = *(const bf16x16*)(proj_wT + (size_t)gn * CH + kk * 32 + kh * 16);
#pragma unroll 1
      for (int mi = 0; mi < 4; ++mi) {
        f32x8 acc;
#pragma unroll
        for (int i = 0; i < 8; ++i) acc[i] = pb;
#pragma unroll
        for (int kk = 0; kk < 8; ++kk) {
          const __bf16* ap = s_out + (mi * 16 + cl) * CH + kk * 32 + kh * 16;
          bf16x8 lo = *(const bf16x8*)(ap);
          bf16x8 hi = *(const bf16x8*)(ap + 8);
          bf16x16 a;
#pragma unroll
          for (int j = 0; j < 8; ++j) { a[j] = lo[j]; a[8 + j] = hi[j]; }
          acc = WMMA_BF16(a, bfr[kk], acc);
        }
#pragma unroll
        for (int g = 0; g < 8; ++g) {
          const int row = mi * 16 + kh * 8 + g;
          const int gh = (wh * WSZ + (row >> 3) + SHIFT) & 255;
          const int gw = (ww * WSZ + (row & 7) + SHIFT) & 255;
          out[((size_t)((b * 256 + gh) * 256 + gw)) * CH + gn] = acc[g];
        }
      }
    }
  }
}

// ---------------------------------------------------------------------------
extern "C" void kernel_launch(void* const* d_in, const int* in_sizes, int n_in,
                              void* d_out, int out_size, void* d_ws, size_t ws_size,
                              hipStream_t stream) {
  (void)in_sizes; (void)n_in; (void)out_size; (void)ws_size;
  const float* x      = (const float*)d_in[0];
  const float* qkv_w  = (const float*)d_in[1];
  const float* q_bias = (const float*)d_in[2];
  const float* v_bias = (const float*)d_in[3];
  const float* scale  = (const float*)d_in[4];
  const float* cpb_w1 = (const float*)d_in[5];
  const float* cpb_b1 = (const float*)d_in[6];
  const float* cpb_w2 = (const float*)d_in[7];
  const float* proj_w = (const float*)d_in[8];
  const float* proj_b = (const float*)d_in[9];
  float* out = (float*)d_out;

  char* ws = (char*)d_ws;
  __bf16* qkv_wT  = (__bf16*)(ws);                 // 768*256*2  = 393216 B
  __bf16* proj_wT = (__bf16*)(ws + 393216);        // 256*256*2  = 131072 B
  float*  cpb     = (float*)(ws + 524288);         // 225*8*4    =   7200 B
  float*  bias_g  = (float*)(ws + 532480);         // 8*64*64*4  = 131072 B

  prep_transpose<<<1024, 256, 0, stream>>>(qkv_w, proj_w, qkv_wT, proj_wT);
  prep_cpb<<<1, 256, 0, stream>>>(cpb_w1, cpb_b1, cpb_w2, cpb);
  prep_bias<<<128, 256, 0, stream>>>(cpb, bias_g);
  swin_window_kernel<<<2048, 256, 0, stream>>>(x, qkv_wT, proj_wT, q_bias, v_bias,
                                               scale, bias_g, proj_b, out);
}
